// LightweightIPGLayer_67164698575269
// MI455X (gfx1250) — matrix operand order repaired and checked by reading, not compiled, not run on hardware
//
#include <hip/hip_runtime.h>
#include <math.h>

// Problem constants (from reference): B=8, C=128, H=W=96
#define BB   8
#define CC   128
#define C2   256
#define HH   96
#define WW   96
#define PP   (HH * WW)        // 9216
#define BP   (BB * PP)        // 73728
#define NGRP 16
#define GCH  (CC / NGRP)      // 8 channels per group

typedef __attribute__((ext_vector_type(16))) _Float16 v16h;
typedef __attribute__((ext_vector_type(8)))  float    v8f;

// ---------------------------------------------------------------------------
// Kernel 0: init per-batch min/max slots (bit-ordered unsigned for nonneg f32)
// ---------------------------------------------------------------------------
__global__ void k_init(unsigned* mm) {
    int t = threadIdx.x;
    if (t < BB) {
        mm[t * 2 + 0] = 0x7F800000u;  // +inf  (min slot)
        mm[t * 2 + 1] = 0x00000000u;  // 0     (max slot)
    }
}

// ---------------------------------------------------------------------------
// Kernel 1: convert FFN weights f32 -> f16
// ---------------------------------------------------------------------------
__global__ void k_convertw(const float* __restrict__ w1, const float* __restrict__ w2,
                           _Float16* __restrict__ w1h, _Float16* __restrict__ w2h) {
    int i = blockIdx.x * 256 + threadIdx.x;   // 0 .. 32767
    if (i < C2 * CC) {
        w1h[i] = (_Float16)w1[i];
        w2h[i] = (_Float16)w2[i];
    }
}

// ---------------------------------------------------------------------------
// Kernel 2: per-pixel channel mean (for DF) and channel L2 norm (for cosine)
// ---------------------------------------------------------------------------
__global__ __launch_bounds__(256) void k_meannorm(const float* __restrict__ x,
                                                  float* __restrict__ avg,
                                                  float* __restrict__ nrm) {
    int gid = blockIdx.x * 256 + threadIdx.x;        // over B*P
    int b = gid / PP, p = gid % PP;
    const float* xp = x + (size_t)b * CC * PP + p;
    float s = 0.f, s2 = 0.f;
    #pragma unroll 4
    for (int c = 0; c < CC; ++c) {
        float v = xp[(size_t)c * PP];
        s += v; s2 += v * v;
    }
    avg[gid] = s * (1.0f / CC);
    nrm[gid] = fmaxf(sqrtf(s2), 1e-12f);             // F.normalize eps
}

// ---------------------------------------------------------------------------
// Kernel 3: Laplacian (zero pad) |.| + per-batch min/max via block reduce+atomics
// ---------------------------------------------------------------------------
__global__ __launch_bounds__(256) void k_df(const float* __restrict__ avg,
                                            float* __restrict__ df,
                                            unsigned* __restrict__ mm) {
    int gid = blockIdx.x * 256 + threadIdx.x;
    int b = gid / PP, p = gid % PP;
    int y = p / WW, xx = p % WW;
    const float* a = avg + (size_t)b * PP;
    float c  = a[p];
    float up = (y  > 0     ) ? a[p - WW] : 0.f;
    float dn = (y  < HH - 1) ? a[p + WW] : 0.f;
    float lf = (xx > 0     ) ? a[p - 1 ] : 0.f;
    float rt = (xx < WW - 1) ? a[p + 1 ] : 0.f;
    float v  = fabsf(4.f * c - up - dn - lf - rt);
    df[gid] = v;

    __shared__ float smin[256], smax[256];
    int t = threadIdx.x;
    smin[t] = v; smax[t] = v;
    __syncthreads();
    for (int s = 128; s > 0; s >>= 1) {
        if (t < s) {
            smin[t] = fminf(smin[t], smin[t + s]);
            smax[t] = fmaxf(smax[t], smax[t + s]);
        }
        __syncthreads();
    }
    if (t == 0) {  // P % 256 == 0 -> block fully inside one batch
        atomicMin(&mm[b * 2 + 0], __float_as_uint(smin[0]));
        atomicMax(&mm[b * 2 + 1], __float_as_uint(smax[0]));
    }
}

// ---------------------------------------------------------------------------
// Kernel 4: GroupNorm statistics: one WG per (b, group)
// ---------------------------------------------------------------------------
__global__ __launch_bounds__(256) void k_gnstats(const float* __restrict__ x,
                                                 float* __restrict__ mu,
                                                 float* __restrict__ rinv) {
    int bg = blockIdx.x;                  // 0 .. B*NGRP-1
    int b = bg / NGRP, g = bg % NGRP;
    const float* xp = x + (size_t)b * CC * PP + (size_t)g * GCH * PP;
    const int N = GCH * PP;               // 73728
    float s = 0.f, s2 = 0.f;
    for (int i = threadIdx.x; i < N; i += 256) {
        float v = xp[i];
        s += v; s2 += v * v;
    }
    __shared__ float r1[256], r2[256];
    int t = threadIdx.x;
    r1[t] = s; r2[t] = s2;
    __syncthreads();
    for (int st = 128; st > 0; st >>= 1) {
        if (t < st) { r1[t] += r1[t + st]; r2[t] += r2[t + st]; }
        __syncthreads();
    }
    if (t == 0) {
        float invN = 1.0f / (float)N;
        float m = r1[0] * invN;
        float var = r2[0] * invN - m * m;  // biased (matches torch GroupNorm)
        mu[bg]   = m;
        rinv[bg] = rsqrtf(var + 1e-5f);
    }
}

__device__ __forceinline__ int reflect96(int i) {
    if (i < 0)    i = -i;            // -1 -> 1
    if (i >= 96)  i = 190 - i;       // 96 -> 94
    return i;
}

// ---------------------------------------------------------------------------
// Kernel 5: per-pixel cosine top-k softmax aggregation + GroupNorm -> enhanced
// Also (optionally) emits an f16 copy of enhanced in [b][p][c] layout so the
// FFN kernel can stage it with async global->LDS copies.
// ---------------------------------------------------------------------------
__global__ __launch_bounds__(256) void k_main(const float* __restrict__ x,
                                              const float* __restrict__ df,
                                              const unsigned* __restrict__ mm,
                                              const float* __restrict__ nrm,
                                              const float* __restrict__ mu,
                                              const float* __restrict__ rinv,
                                              const float* __restrict__ gw,
                                              const float* __restrict__ gb,
                                              float* __restrict__ out,
                                              _Float16* __restrict__ ef16) {
    int gid = blockIdx.x * 256 + threadIdx.x;  // over B*P
    int b = gid / PP, p = gid % PP;
    int y = p / WW, xx = p % WW;

    // connection count k in [1, 8]
    float dmin = __uint_as_float(mm[b * 2 + 0]);
    float dmax = __uint_as_float(mm[b * 2 + 1]);
    float dn = (df[gid] - dmin) / (dmax - dmin + 1e-8f);
    int kcnt = 1 + (int)rintf(dn * 7.f);       // round-half-even like jnp.round
    if (kcnt < 1) kcnt = 1;
    if (kcnt > 9) kcnt = 9;

    // 3x3 reflect-padded neighbor pixel offsets
    int q[9];
    #pragma unroll
    for (int dy = 0; dy < 3; ++dy)
        #pragma unroll
        for (int dx = 0; dx < 3; ++dx)
            q[dy * 3 + dx] = reflect96(y + dy - 1) * WW + reflect96(xx + dx - 1);

    // channel dot products center . neighbor
    const float* xb = x + (size_t)b * CC * PP;
    float dot[9];
    #pragma unroll
    for (int j = 0; j < 9; ++j) dot[j] = 0.f;
    for (int c = 0; c < CC; ++c) {
        const float* xc = xb + (size_t)c * PP;
        float cv = xc[p];
        #pragma unroll
        for (int j = 0; j < 9; ++j) dot[j] += cv * xc[q[j]];
    }

    const float* nb = nrm + (size_t)b * PP;
    float np0 = nb[p];
    float sim[9];
    #pragma unroll
    for (int j = 0; j < 9; ++j) sim[j] = dot[j] / (np0 * nb[q[j]]);

    // stable insertion sort, descending (matches argsort(-sim) tie order)
    float ss[9]; int ord[9];
    #pragma unroll
    for (int j = 0; j < 9; ++j) {
        float v = sim[j]; int pos = j;
        while (pos > 0 && ss[pos - 1] < v) {
            ss[pos] = ss[pos - 1]; ord[pos] = ord[pos - 1]; --pos;
        }
        ss[pos] = v; ord[pos] = j;
    }

    // softmax over the top-k ranked sims
    float wts[9]; float wsum = 0.f;
    #pragma unroll
    for (int r = 0; r < 9; ++r) {
        float w = (r < kcnt) ? expf(ss[r]) : 0.f;
        wts[r] = w; wsum += w;
    }
    float winv = 1.0f / wsum;
    #pragma unroll
    for (int r = 0; r < 9; ++r) wts[r] *= winv;

    // aggregate + GroupNorm -> enhanced (written to d_out [+ f16 copy])
    float* ob = out + (size_t)b * CC * PP;
    const float* mub = mu + b * NGRP;
    const float* rib = rinv + b * NGRP;
    _Float16* eb = ef16 ? (ef16 + (size_t)gid * CC) : nullptr;  // [b][p][c]
    for (int c = 0; c < CC; ++c) {
        const float* xc = xb + (size_t)c * PP;
        float acc = 0.f;
        #pragma unroll
        for (int r = 0; r < 9; ++r)
            if (r < kcnt) acc += wts[r] * xc[q[ord[r]]];
        int g = c >> 3;
        float e = acc + (xc[p] - mub[g]) * rib[g] * gw[c] + gb[c];
        ob[(size_t)c * PP + p] = e;
        if (eb) eb[c] = (_Float16)e;
    }
}

// ---------------------------------------------------------------------------
// Kernel 6: fused 2-layer FFN (1x1 convs) with WMMA, in-place out += ffn(out)
//   layer1: h = relu(w1[256x128] * e + b1),  layer2: out += w2[128x256] * h + b2
// WG = 256 threads (8 waves), tile = 64 pixels, one batch each.
// ASYNC=true stages the activation tile with GLOBAL_LOAD_ASYNC_TO_LDS_B128
// from the f16 [b][p][c] copy (ASYNCcnt-tracked, padded LDS rows).
// ---------------------------------------------------------------------------
#define TN 64
template <bool ASYNC>
__global__ __launch_bounds__(256) void k_ffn(float* __restrict__ out,
                                             const _Float16* __restrict__ ef16,
                                             const _Float16* __restrict__ w1h,
                                             const float* __restrict__ b1,
                                             const _Float16* __restrict__ w2h,
                                             const float* __restrict__ b2) {
    const int tilesPerB = PP / TN;                 // 144
    int b  = blockIdx.x / tilesPerB;
    int p0 = (blockIdx.x % tilesPerB) * TN;

    __shared__ __align__(16) _Float16 Xs[TN][CC + 8];   // enhanced tile [n][k]
    __shared__ __align__(16) _Float16 Hs[TN][C2 + 8];   // hidden tile   [n][k]

    if (ASYNC) {
        // Per-lane async global->LDS: 1024 chunks of 16B (64 rows x 16 chunks).
        // Source rows are contiguous 256B in ef16; LDS rows have +16B padding,
        // inserted by computing each chunk's LDS offset explicitly.
        // ISA 10.2: low 32 bits of a generic LDS pointer are the LDS byte offset.
        const char* src = (const char*)(ef16 + ((size_t)b * PP + p0) * CC);
        unsigned ldsbase = (unsigned)(size_t)&Xs[0][0];
        for (int i = threadIdx.x; i < TN * 16; i += 256) {   // exactly 4 iters, no divergence
            int n = i >> 4, c16 = i & 15;
            unsigned laddr = ldsbase + (unsigned)(n * (CC + 8) + c16 * 8) * 2;
            const void* g = src + i * 16;
            asm volatile("global_load_async_to_lds_b128 %0, %1, off"
                         :: "v"(laddr), "v"(g) : "memory");
        }
        asm volatile("s_wait_asynccnt 0x0" ::: "memory");
    } else {
        // Sync fallback: read enhanced f32 from d_out, convert to f16 in LDS.
        const float* xin = out + (size_t)b * CC * PP;
        for (int i = threadIdx.x; i < TN * CC; i += 256) {
            int n = i & (TN - 1);
            int c = i >> 6;
            Xs[n][c] = (_Float16)xin[(size_t)c * PP + p0 + n];
        }
    }
    __syncthreads();

    const int wave = threadIdx.x >> 5;
    const int lane = threadIdx.x & 31;
    const int lm   = lane & 15;                    // A row / B col / D col
    const int h    = lane >> 4;

    // fragment K index maps (CDNA5 ISA 7.12.2, wave32)
    int kA[16], kB[16];
    #pragma unroll
    for (int e = 0; e < 16; ++e) {
        kA[e] = h * 16 + e;                                 // contiguous K order, matches B
        kB[e] = h * 16 + e;                                 // contiguous K order, matches A
    }

    // ---------------- layer 1: M=256, K=128, N=64 ----------------
    // wave owns 2 M-tiles x 4 N-tiles
    for (int mt = wave * 2; mt < wave * 2 + 2; ++mt) {
        const _Float16* arowb = w1h + (size_t)(mt * 16 + lm) * CC;
        for (int nt = 0; nt < 4; ++nt) {
            v8f acc = {};
            #pragma unroll
            for (int ks = 0; ks < 4; ++ks) {       // K = 4 * 32
                int k0 = ks * 32;
                v16h af, bf;
                #pragma unroll
                for (int e = 0; e < 16; ++e) {
                    af[e] = arowb[k0 + kA[e]];
                    bf[e] = Xs[nt * 16 + lm][k0 + kB[e]];
                }
                acc = __builtin_amdgcn_wmma_f32_16x16x32_f16(
                          false, af, false, bf, (short)0, acc, false, false);
            }
            #pragma unroll
            for (int r = 0; r < 8; ++r) {          // bias + ReLU -> Hs (f16)
                int m = mt * 16 + r + 8 * h;
                float v = acc[r] + b1[m];
                Hs[nt * 16 + lm][m] = (_Float16)(v > 0.f ? v : 0.f);
            }
        }
    }
    __syncthreads();

    // ---------------- layer 2: M=128, K=256, N=64 ----------------
    // wave owns 1 M-tile x 4 N-tiles; out += w2*h + b2 (in place over enhanced)
    {
        int mt = wave;
        const _Float16* arowb = w2h + (size_t)(mt * 16 + lm) * C2;
        for (int nt = 0; nt < 4; ++nt) {
            v8f acc = {};
            #pragma unroll
            for (int ks = 0; ks < 8; ++ks) {       // K = 8 * 32
                int k0 = ks * 32;
                v16h af, bf;
                #pragma unroll
                for (int e = 0; e < 16; ++e) {
                    af[e] = arowb[k0 + kA[e]];
                    bf[e] = Hs[nt * 16 + lm][k0 + kB[e]];
                }
                acc = __builtin_amdgcn_wmma_f32_16x16x32_f16(
                          false, af, false, bf, (short)0, acc, false, false);
            }
            #pragma unroll
            for (int r = 0; r < 8; ++r) {
                int m = mt * 16 + r + 8 * h;
                size_t idx = (size_t)b * CC * PP + (size_t)m * PP + p0 + nt * 16 + lm;
                out[idx] = out[idx] + acc[r] + b2[m];
            }
        }
    }
}

// ---------------------------------------------------------------------------
// Launch
// ---------------------------------------------------------------------------
extern "C" void kernel_launch(void* const* d_in, const int* in_sizes, int n_in,
                              void* d_out, int out_size, void* d_ws, size_t ws_size,
                              hipStream_t stream) {
    const float* x   = (const float*)d_in[0];   // [B,C,H,W]
    const float* gw  = (const float*)d_in[1];   // [C]
    const float* gb  = (const float*)d_in[2];   // [C]
    const float* w1  = (const float*)d_in[3];   // [2C, C]
    const float* b1  = (const float*)d_in[4];   // [2C]
    const float* w2  = (const float*)d_in[5];   // [C, 2C]
    const float* b2  = (const float*)d_in[6];   // [C]
    float* out = (float*)d_out;

    // workspace layout (~1.02 MB base + optional 18 MB f16 enhanced copy)
    float* ws       = (float*)d_ws;
    float* avg      = ws;                         // BP
    float* nrm      = avg + BP;                   // BP
    float* dfb      = nrm + BP;                   // BP
    unsigned* mm    = (unsigned*)(dfb + BP);      // B*2 (padded to 16)
    float* mu       = (float*)(mm + 2 * BB);      // B*16
    float* rinv     = mu + BB * NGRP;             // B*16
    _Float16* w1h   = (_Float16*)(rinv + BB * NGRP);  // 256*128
    _Float16* w2h   = w1h + C2 * CC;                  // 128*256
    _Float16* ef16  = w2h + CC * C2;                  // BP*CC (optional)

    size_t need_async = (size_t)((char*)(ef16 + (size_t)BP * CC) - (char*)d_ws);
    bool useAsync = (ws_size >= need_async);      // constant per run -> deterministic

    (void)in_sizes; (void)n_in; (void)out_size;

    k_init<<<1, 32, 0, stream>>>(mm);
    k_convertw<<<(C2 * CC + 255) / 256, 256, 0, stream>>>(w1, w2, w1h, w2h);
    k_meannorm<<<BP / 256, 256, 0, stream>>>(x, avg, nrm);
    k_df<<<BP / 256, 256, 0, stream>>>(avg, dfb, mm);
    k_gnstats<<<BB * NGRP, 256, 0, stream>>>(x, mu, rinv);
    k_main<<<BP / 256, 256, 0, stream>>>(x, dfb, mm, nrm, mu, rinv, gw, gb, out,
                                         useAsync ? ef16 : (_Float16*)nullptr);
    if (useAsync) {
        k_ffn<true><<<BB * (PP / TN), 256, 0, stream>>>(out, ef16, w1h, b1, w2h, b2);
    } else {
        k_ffn<false><<<BB * (PP / TN), 256, 0, stream>>>(out, nullptr, w1h, b1, w2h, b2);
    }
}
